// LocalLinear_4011499455027
// MI455X (gfx1250) — matrix operand order repaired and checked by reading, not compiled, not run on hardware
//
#include <hip/hip_runtime.h>

// Problem constants (match reference)
#define IN_FEATURES     4096
#define LOCAL_FEATURES  128
#define KERNEL_SIZE     64
#define STRIDE_LEN      32
#define FOLD_NUM        127               // (4096-64)/32 + 1
#define BATCH           4096
#define OUT_W           (FOLD_NUM * LOCAL_FEATURES)   // 16256

// Tiling
#define M_TILE  128     // rows of x per block (amortizes W[f] conversion)
#define XS      72      // bf16 row stride: 144B = 16B-aligned, 36-bank stride -> conflict-free 16B gathers

typedef __attribute__((ext_vector_type(16))) __bf16 v16bf;
typedef __attribute__((ext_vector_type(8)))  __bf16 v8bf;
typedef __attribute__((ext_vector_type(8)))  float  v8f;

// ---- f32 -> bf16 RNE split helpers (pure bit ops; no target bf16-arith dependency) ----
__device__ __forceinline__ unsigned rne_bits(float v) {
    unsigned u = __builtin_bit_cast(unsigned, v);
    return u + 0x7FFFu + ((u >> 16) & 1u);     // rounded; bf16 = high 16 bits
}
// Split two consecutive f32 into a packed bf16 hi-pair and lo-pair (a -> low half)
__device__ __forceinline__ void split2(float a, float b, unsigned& hp, unsigned& lp) {
    unsigned ra = rne_bits(a), rb = rne_bits(b);
    hp = (ra >> 16) | (rb & 0xFFFF0000u);
    float la = a - __builtin_bit_cast(float, ra & 0xFFFF0000u);
    float lb = b - __builtin_bit_cast(float, rb & 0xFFFF0000u);
    unsigned sa = rne_bits(la), sb = rne_bits(lb);
    lp = (sa >> 16) | (sb & 0xFFFF0000u);
}

// ---- Fragment gathers: pure ds_load_b128 pairs, layouts per ISA 7.12.2 ----
// A (16-bit, 16x32): lane(h,m) element i -> k = c*32 + ((i<8)? i : i+8) + 8h
//   => elements 0-7  : 8 contiguous bf16 at k0 = c*32 + 8h
//      elements 8-15 : 8 contiguous bf16 at k0 + 16
__device__ __forceinline__ v16bf gatherA(const unsigned short* base, int row, int c, int h) {
    const __bf16* rp = (const __bf16*)(base + row * XS);
    v8bf p = *(const v8bf*)(rp + c * 32 + 8 * h);
    v8bf q = *(const v8bf*)(rp + c * 32 + 16 + 8 * h);
    v16bf a;
    #pragma unroll
    for (int i = 0; i < 8; ++i) { a[i] = p[i]; a[i + 8] = q[i]; }
    return a;
}
// B (16-bit, 32x16): lane(h,n) element i -> k = c*32 + ((i<8)? i : i+8) + 8h (same as A)
__device__ __forceinline__ v16bf gatherB(const unsigned short* base, int row, int c, int h) {
    const __bf16* rp = (const __bf16*)(base + row * XS);
    v8bf p = *(const v8bf*)(rp + c * 32 + 8 * h);
    v8bf q = *(const v8bf*)(rp + c * 32 + 16 + 8 * h);
    v16bf b;
    #pragma unroll
    for (int i = 0; i < 8; ++i) { b[i] = p[i]; b[i + 8] = q[i]; }
    return b;
}

__global__ __launch_bounds__(256)
void local_linear_wmma(const float* __restrict__ x,
                       const float* __restrict__ W,
                       const float* __restrict__ bvec,
                       float* __restrict__ out)
{
    // Pre-split bf16 tiles: each element converted exactly once per block.
    __shared__ __align__(16) unsigned short xs_hi[M_TILE * XS];         // 18 KB
    __shared__ __align__(16) unsigned short xs_lo[M_TILE * XS];         // 18 KB
    __shared__ __align__(16) unsigned short ws_hi[LOCAL_FEATURES * XS]; // 18 KB
    __shared__ __align__(16) unsigned short ws_lo[LOCAL_FEATURES * XS]; // 18 KB

    const int f   = blockIdx.x;    // fold, 0..126
    const int mb  = blockIdx.y;    // m-block, 0..31
    const int tid = threadIdx.x;   // 0..255

    // ---- Stage + split x window tile (128 rows x 64 k), coalesced float2 loads ----
    const float* xwin = x + (size_t)mb * M_TILE * IN_FEATURES + (size_t)f * STRIDE_LEN;
    for (int p = tid; p < M_TILE * (KERNEL_SIZE / 2); p += 256) {
        int r = p >> 5, c = (p & 31) * 2;
        const float* s = xwin + (size_t)r * IN_FEATURES + c;
        unsigned hp, lp;
        split2(s[0], s[1], hp, lp);
        *(unsigned*)&xs_hi[r * XS + c] = hp;
        *(unsigned*)&xs_lo[r * XS + c] = lp;
    }
    // ---- Stage + split W[f] (128 rows x 64 k, contiguous) ----
    const float* Wf = W + (size_t)f * LOCAL_FEATURES * KERNEL_SIZE;
    for (int p = tid; p < LOCAL_FEATURES * (KERNEL_SIZE / 2); p += 256) {
        int r = p >> 5, c = (p & 31) * 2;
        const float* s = Wf + r * KERNEL_SIZE + c;
        unsigned hp, lp;
        split2(s[0], s[1], hp, lp);
        *(unsigned*)&ws_hi[r * XS + c] = hp;
        *(unsigned*)&ws_lo[r * XS + c] = lp;
    }
    __syncthreads();

    const int wave = tid >> 5;        // 0..7 -> owns N-tile n0 = wave*16
    const int lane = tid & 31;
    const int h    = lane >> 4;
    const int ln   = lane & 15;
    const int n0   = wave * 16;

    // ---- B fragments (hi/lo), built once, reused across all 8 M-tiles ----
    // B[k][n] = W[f][n][k]: lane's column n = n0+ln is W row n0+ln.
    v16bf bh[2], bl[2];
    #pragma unroll
    for (int c = 0; c < 2; ++c) {
        bh[c] = gatherB(ws_hi, n0 + ln, c, h);
        bl[c] = gatherB(ws_lo, n0 + ln, c, h);
    }

    // bias is per-column -> identical in all 8 C VGPRs of this lane: fold into acc init
    const float bias = bvec[(size_t)f * LOCAL_FEATURES + n0 + ln];
    const size_t col = (size_t)f * LOCAL_FEATURES + n0 + ln;

    for (int mt = 0; mt < M_TILE / 16; ++mt) {
        v16bf ah[2], al[2];
        #pragma unroll
        for (int c = 0; c < 2; ++c) {
            ah[c] = gatherA(xs_hi, mt * 16 + ln, c, h);
            al[c] = gatherA(xs_lo, mt * 16 + ln, c, h);
        }

        v8f acc;
        #pragma unroll
        for (int r = 0; r < 8; ++r) acc[r] = bias;

        // bf16x3 split: xh*wh + xh*wl + xl*wh, f32 accumulate
        #pragma unroll
        for (int c = 0; c < 2; ++c) {
            acc = __builtin_amdgcn_wmma_f32_16x16x32_bf16(false, ah[c], false, bh[c],
                                                          (short)0, acc, false, false);
            acc = __builtin_amdgcn_wmma_f32_16x16x32_bf16(false, ah[c], false, bl[c],
                                                          (short)0, acc, false, false);
            acc = __builtin_amdgcn_wmma_f32_16x16x32_bf16(false, al[c], false, bh[c],
                                                          (short)0, acc, false, false);
        }

        // C/D layout: VGPR r -> row (mt*16 + 8h + r), col = n0+ln.
        // Output is a 266MB zero-reuse stream: store non-temporal to protect L2-resident x/W.
        const int mrow = mb * M_TILE + mt * 16 + 8 * h;
        #pragma unroll
        for (int r = 0; r < 8; ++r) {
            __builtin_nontemporal_store(acc[r], &out[(size_t)(mrow + r) * OUT_W + col]);
        }
    }
}

extern "C" void kernel_launch(void* const* d_in, const int* in_sizes, int n_in,
                              void* d_out, int out_size, void* d_ws, size_t ws_size,
                              hipStream_t stream) {
    const float* x   = (const float*)d_in[0];   // (4096, 4096) f32
    const float* W   = (const float*)d_in[1];   // (127, 128, 64) f32
    const float* b   = (const float*)d_in[2];   // (127, 128) f32
    float*       out = (float*)d_out;           // (4096, 16256) f32

    dim3 grid(FOLD_NUM, BATCH / M_TILE);        // 127 x 32 blocks
    dim3 block(256);                            // 8 wave32s
    local_linear_wmma<<<grid, block, 0, stream>>>(x, W, b, out);
}
